// SQN_81484119540477
// MI455X (gfx1250) — matrix operand order repaired and checked by reading, not compiled, not run on hardware
//
#include <hip/hip_runtime.h>

// SNN forward: 3-layer LIF, 100 steps. Persistent neuron state and the wave's
// W1 slice live in VGPRs for the entire simulation; spike matmuls via
// v_wmma_f32_16x16x32_f16 accumulating directly into syn registers.
// 8 waves x 32-col slices over a 16-row tile => ~230 VGPRs/wave (no spills).

typedef __attribute__((ext_vector_type(16))) _Float16 v16h;
typedef __attribute__((ext_vector_type(8)))  _Float16 v8h;
typedef __attribute__((ext_vector_type(8)))  float    v8f;

#define ALPHA 0.9f
#define BETA  0.85f
#define SIM_T 100

#define B_TOTAL 32768
#define N_IN    128
#define N_HID   256
#define N_OUT   8
#define ROWS_PER_WG 16
#define THREADS 256

__global__ __launch_bounds__(THREADS, 1)
void snn_fused_kernel(const float* __restrict__ inp,
                      const float* __restrict__ W0,
                      const float* __restrict__ W1,
                      const float* __restrict__ W2,
                      float* __restrict__ out)
{
    // LDS: 128K + 8K + 8K + 8K = 152 KB (<= 320 KB WGP; 2 WGs can co-reside)
    __shared__ _Float16 sW1T[N_HID * N_HID];        // [col][k] f16 (prologue only)
    __shared__ _Float16 sW2T[16 * N_HID];           // [col][k] f16, cols 8..15 = 0
    __shared__ _Float16 sOut0[ROWS_PER_WG * N_HID]; // layer-0 spikes, row-major f16
    __shared__ _Float16 sOut1[ROWS_PER_WG * N_HID]; // layer-1 spikes, row-major f16

    const int tid   = threadIdx.x;
    const int lane  = tid & 31;
    const int wave  = tid >> 5;      // 0..7 : which 32-neuron slice
    const int wgRow = blockIdx.x * ROWS_PER_WG;
    // wave-uniform predicate as an SGPR value -> scalar branch, EXEC untouched
    const bool isOutWave = (__builtin_amdgcn_readfirstlane(wave) == 0);

    // ---------- stage W1^T (f16) and padded W2^T into LDS ----------
    {
        const int k = tid; // 0..255
        for (int n = 0; n < N_HID; ++n)
            sW1T[n * N_HID + k] = (_Float16)W1[k * N_HID + n];
        for (int c = 0; c < 16; ++c)
            sW2T[c * N_HID + k] = (c < N_OUT) ? (_Float16)W2[k * N_OUT + c]
                                              : (_Float16)0.0f;
    }
    // reuse sOut0 region as f32 input staging [16][128] (8 KB) for the prologue
    float* sInp = (float*)sOut0;
    for (int i = tid; i < ROWS_PER_WG * N_IN; i += THREADS)
        sInp[i] = inp[(wgRow + (i >> 7)) * N_IN + (i & 127)];
    __syncthreads();

    // ---------- per-lane C-layout geometry ----------
    const int hi      = (lane >> 4) & 1;         // 0: rows v+0..7, 1: rows v+8..15
    const int mloc    = hi * 8;                  // local row base (add v)
    const int ccol    = lane & 15;               // column within 16-wide tile
    const int colBase = wave * 32;               // this wave's neuron slice

    // A/B fragment geometry (documented 16-bit WMMA VGPR layouts)
    const int arow = lane & 15;                  // A: both half-waves -> M=0..15
    const int asel = hi * 8;                     // A lane<16: K{0..7,16..23} else {8..15,24..31}
    const int bcol = lane & 15;                  // B: column per lane
    const int bsel = hi * 8;                     // B lane<16: K{0..7,16..23} else {8..15,24..31}

    // ---------- persistent state (registers) ----------
    v8f h0[2], mem0[2], syn1[2], mem1[2];
    const v8f vzero = {};

    // h0 = inp @ W0 computed once, exact f32 FMA, directly into C layout
    for (int nt = 0; nt < 2; ++nt) {
        const int col = colBase + nt * 16 + ccol;
        v8f acc = vzero;
        for (int k = 0; k < N_IN; ++k) {
            const float w = W0[k * N_HID + col];
            #pragma unroll
            for (int v = 0; v < 8; ++v)
                acc[v] += sInp[(mloc + v) * N_IN + k] * w;
        }
        h0[nt]   = acc;
        mem0[nt] = vzero;
        syn1[nt] = vzero;
        mem1[nt] = vzero;
    }
    v8f syn2 = vzero, mem2 = vzero;
    float c0 = 0.0f;                 // syn0 == h0 * c0 (closed form, no reset path)
    __syncthreads();                 // done using sInp region

    // ---------- hoist this wave's W1 slice into registers ----------
    // 2 n-tiles x 8 k-chunks x v16h = 128 VGPRs, loop-invariant for 100 steps.
    union AB { v16h v; v8h h[2]; };
    v16h w1frag[2][8];
    #pragma unroll
    for (int nt = 0; nt < 2; ++nt) {
        #pragma unroll
        for (int kc = 0; kc < 8; ++kc) {
            AB Bf;
            const _Float16* bp =
                &sW1T[(colBase + nt * 16 + bcol) * N_HID + kc * 32 + bsel];
            Bf.h[0] = *(const v8h*)(bp);
            Bf.h[1] = *(const v8h*)(bp + 16);
            w1frag[nt][kc] = Bf.v;
        }
    }

    #pragma unroll 1
    for (int t = 0; t < SIM_T; ++t) {
        // ---- phase 1: spikes from OLD mem, membrane decay (pre-GEMM state) ----
        #pragma unroll
        for (int nt = 0; nt < 2; ++nt) {
            const int col = colBase + nt * 16 + ccol;
            v8f m0 = mem0[nt], m1 = mem1[nt];
            #pragma unroll
            for (int v = 0; v < 8; ++v) {
                const float s0 = (m0[v] > 1.0f) ? 1.0f : 0.0f;   // spike / reset L0
                const float s1 = (m1[v] > 1.0f) ? 1.0f : 0.0f;   // spike / reset L1
                sOut0[(mloc + v) * N_HID + col] = (_Float16)s0;
                sOut1[(mloc + v) * N_HID + col] = (_Float16)s1;
                m0[v] = BETA * m0[v] + h0[nt][v] * c0 - s0;      // uses OLD syn0
                m1[v] = BETA * m1[v] + syn1[nt][v] - s1;         // uses OLD syn1
            }
            mem0[nt] = m0;
            mem1[nt] = m1;
        }
        if (isOutWave) {            // scalar branch
            #pragma unroll
            for (int v = 0; v < 8; ++v)
                mem2[v] = BETA * mem2[v] + syn2[v];              // uses OLD syn2
        }
        c0 = ALPHA * c0 + 1.0f;
        __syncthreads();

        // ---- phase 2: syn1 = alpha*syn1 + out0 @ W1 (WMMA accumulates in place) ----
        #pragma unroll
        for (int nt = 0; nt < 2; ++nt) {
            #pragma unroll
            for (int v = 0; v < 8; ++v)
                syn1[nt][v] *= ALPHA;
        }
        #pragma unroll
        for (int kc = 0; kc < 8; ++kc) {
            AB A;
            const _Float16* ap = &sOut0[arow * N_HID + kc * 32 + asel];
            A.h[0] = *(const v8h*)(ap);
            A.h[1] = *(const v8h*)(ap + 16);
            #pragma unroll
            for (int nt = 0; nt < 2; ++nt) {
                syn1[nt] = __builtin_amdgcn_wmma_f32_16x16x32_f16(
                    false, A.v, false, w1frag[nt][kc], (short)0, syn1[nt],
                    false, false);
            }
        }

        // ---- phase 2b: syn2 = alpha*syn2 + out1 @ W2 (wave 0 only) ----
        if (isOutWave) {            // scalar branch, EXEC stays all-ones
            #pragma unroll
            for (int v = 0; v < 8; ++v)
                syn2[v] *= ALPHA;
            #pragma unroll
            for (int kc = 0; kc < 8; ++kc) {
                AB A2, B2;
                const _Float16* a2p = &sOut1[arow * N_HID + kc * 32 + asel];
                A2.h[0] = *(const v8h*)(a2p);
                A2.h[1] = *(const v8h*)(a2p + 16);
                const _Float16* b2p = &sW2T[bcol * N_HID + kc * 32 + bsel];
                B2.h[0] = *(const v8h*)(b2p);
                B2.h[1] = *(const v8h*)(b2p + 16);
                syn2 = __builtin_amdgcn_wmma_f32_16x16x32_f16(
                    false, A2.v, false, B2.v, (short)0, syn2, false, false);
            }
        }
        __syncthreads();   // protect spike staging before next step's writes
    }

    // ---- final: decode = output-layer membrane potential ----
    if (isOutWave && ccol < N_OUT) {
        #pragma unroll
        for (int v = 0; v < 8; ++v)
            out[(wgRow + mloc + v) * N_OUT + ccol] = mem2[v];
    }
}

extern "C" void kernel_launch(void* const* d_in, const int* in_sizes, int n_in,
                              void* d_out, int out_size, void* d_ws, size_t ws_size,
                              hipStream_t stream) {
    (void)in_sizes; (void)n_in; (void)out_size; (void)d_ws; (void)ws_size;
    const float* inp = (const float*)d_in[0];   // [32768,128]
    const float* W0  = (const float*)d_in[1];   // [128,256]
    const float* W1  = (const float*)d_in[2];   // [256,256]
    const float* W2  = (const float*)d_in[3];   // [256,8]
    float* out = (float*)d_out;                 // [32768,8]
    snn_fused_kernel<<<dim3(B_TOTAL / ROWS_PER_WG), dim3(THREADS), 0, stream>>>(
        inp, W0, W1, W2, out);
}
